// Decoder_51144470561483
// MI455X (gfx1250) — compile-verified, matrix operand order checked by reading, not run on hardware
//
#include <hip/hip_runtime.h>

#define NUM_H3 5882
#define NUM_LATLON 64800
#define DEG 7
#define NUM_EDGES (NUM_LATLON * DEG)
#define EPS 1e-5f

typedef __attribute__((ext_vector_type(16))) __bf16 v16bf;
typedef __attribute__((ext_vector_type(8)))  __bf16 bf16x8;
typedef __attribute__((ext_vector_type(4)))  __bf16 bf16x4;
typedef __attribute__((ext_vector_type(8)))  float  v8f;
typedef __attribute__((ext_vector_type(4)))  float  f32x4;

// ---------------------------------------------------------------------------
// Fragment load for 16-bit 16x32 A/B tiles (CDNA5 WMMA layout):
// lane L: row = L&15, half = L>>4; elements 0..7  = row[k0 + 8*half .. +7]
//                                  elements 8..15 = row[k0 + 16 + 8*half .. +7]
// Works for A (rows = M) and for B when weights are stored transposed
// (Wt[N][K] row-major, rows = N).
// ---------------------------------------------------------------------------
__device__ __forceinline__ v16bf load_frag(const __bf16* base, int ld) {
  const int lane = threadIdx.x & 31;
  const int r = lane & 15;
  const int h = lane >> 4;
  const __bf16* p = base + r * ld + h * 8;
  bf16x8 lo = *(const bf16x8*)(p);
  bf16x8 hi = *(const bf16x8*)(p + 16);
  v16bf f;
#pragma unroll
  for (int i = 0; i < 8; ++i) { f[i] = lo[i]; f[i + 8] = hi[i]; }
  return f;
}

__device__ __forceinline__ v8f wmma_bf16(v8f acc, v16bf a, v16bf b) {
  return __builtin_amdgcn_wmma_f32_16x16x32_bf16(false, a, false, b,
                                                 (short)0, acc, false, false);
}

// acc += A(16 x K, row-major lda) * Wt(nt-th 16-col block; Wt is [N][K])
__device__ __forceinline__ v8f gemm_acc(v8f acc, const __bf16* A, int lda,
                                        const __bf16* Wt, int ldw,
                                        int nt, int kTiles) {
  for (int kt = 0; kt < kTiles; ++kt) {
    v16bf a = load_frag(A + kt * 32, lda);
    v16bf b = load_frag(Wt + nt * 16 * ldw + kt * 32, ldw);
    acc = wmma_bf16(acc, a, b);
  }
  return acc;
}

// 4-way-ILP dot of a 256-long bf16 LDS row against a strided f32 column
// Wcol[k*2 + o]; used for the 256->2 heads.
__device__ __forceinline__ float dot256_ln2(const __bf16* row,
                                            const float* W2, int o) {
  float a0 = 0.f, a1 = 0.f, a2 = 0.f, a3 = 0.f;
  for (int k = 0; k < 256; k += 8) {
    bf16x8 xv = *(const bf16x8*)(row + k);
    a0 = fmaf((float)xv[0], W2[(k + 0) * 2 + o], a0);
    a1 = fmaf((float)xv[1], W2[(k + 1) * 2 + o], a1);
    a2 = fmaf((float)xv[2], W2[(k + 2) * 2 + o], a2);
    a3 = fmaf((float)xv[3], W2[(k + 3) * 2 + o], a3);
    a0 = fmaf((float)xv[4], W2[(k + 4) * 2 + o], a0);
    a1 = fmaf((float)xv[5], W2[(k + 5) * 2 + o], a1);
    a2 = fmaf((float)xv[6], W2[(k + 6) * 2 + o], a2);
    a3 = fmaf((float)xv[7], W2[(k + 7) * 2 + o], a3);
  }
  return (a0 + a1) + (a2 + a3);
}

// ---------------------------------------------------------------------------
// Weight prep: Wt[n][k] = (bf16) W[k][n], rows n >= N zero-padded.
// ---------------------------------------------------------------------------
__global__ void transpose_cast_kernel(const float* __restrict__ W,
                                      __bf16* __restrict__ Wt,
                                      int K, int N, int Npad) {
  int t = blockIdx.x * blockDim.x + threadIdx.x;
  if (t >= Npad * K) return;
  int n = t / K, k = t - n * K;
  float v = (n < N) ? W[(size_t)k * N + n] : 0.0f;
  Wt[(size_t)n * K + k] = (__bf16)v;
}

// ---------------------------------------------------------------------------
// Edge pipeline: edge_enc MLP -> ea ; proc_edge MLP -> e  (written to e_out)
// ---------------------------------------------------------------------------
struct EdgeParams {
  const float* proc;    // processor_features (NUM_H3,256)
  const float* latlon;  // latlon_nodes (NUM_LATLON,256)
  const int*   eidx;    // edge_index (2,E)
  const float* eattr;   // (E,2)
  const float* ee_W0;   // (2,256)
  const float* ee_b0; const float* ee_b1;
  const float* ee_W2;   // (256,2)
  const float* ee_b2; const float* ee_g; const float* ee_beta;
  const float* pe_W0;   // (514,256) f32 (rows 512/513 used here)
  const float* pe_b0; const float* pe_b1;
  const float* pe_W2;   // (256,2)
  const float* pe_b2; const float* pe_g; const float* pe_beta;
  const __bf16* ee_W1t;   // 256x256
  const __bf16* pe_W0at;  // 256x256 (K = src rows)
  const __bf16* pe_W0bt;  // 256x256 (K = dst rows)
  const __bf16* pe_W1t;   // 256x256
  float* e_out;           // (E,2)
};

#define EWAVES 4
__global__ __launch_bounds__(EWAVES * 32) void edge_kernel(EdgeParams P) {
  __shared__ __bf16 sA[EWAVES * 16 * 256];
  __shared__ __bf16 sB[EWAVES * 16 * 256];
  __shared__ __bf16 sC[EWAVES * 16 * 256];
  __shared__ float  sEA[EWAVES * 16 * 2];

  const int lane = threadIdx.x & 31;
  const int wave = threadIdx.x >> 5;
  const int tile = blockIdx.x * EWAVES + wave;
  const bool active = (tile * 16) < NUM_EDGES;   // wave-uniform
  const int eBase = tile * 16;
  __bf16* A = sA + wave * 16 * 256;
  __bf16* B = sB + wave * 16 * 256;
  __bf16* C = sC + wave * 16 * 256;
  float*  EA = sEA + wave * 32;
  const int col = lane & 15, hh = lane >> 4;

  // P1: edge_enc layer0 (K=2, VALU) -> A.  8 contiguous cols per lane,
  // one ds_store_b128 per row; layer-0 weights hoisted out of the row loop.
  if (active) {
    const int nb = lane * 8;
    f32x4 wA0 = *(const f32x4*)(P.ee_W0 + nb);
    f32x4 wA1 = *(const f32x4*)(P.ee_W0 + nb + 4);
    f32x4 wB0 = *(const f32x4*)(P.ee_W0 + 256 + nb);
    f32x4 wB1 = *(const f32x4*)(P.ee_W0 + 256 + nb + 4);
    f32x4 bb0 = *(const f32x4*)(P.ee_b0 + nb);
    f32x4 bb1 = *(const f32x4*)(P.ee_b0 + nb + 4);
    for (int r = 0; r < 16; ++r) {
      float s0 = P.eattr[(size_t)(eBase + r) * 2 + 0];
      float s1 = P.eattr[(size_t)(eBase + r) * 2 + 1];
      bf16x8 pk;
#pragma unroll
      for (int i = 0; i < 4; ++i) {
        pk[i]     = (__bf16)fmaxf(fmaf(s0, wA0[i], fmaf(s1, wB0[i], bb0[i])), 0.0f);
        pk[i + 4] = (__bf16)fmaxf(fmaf(s0, wA1[i], fmaf(s1, wB1[i], bb1[i])), 0.0f);
      }
      *(bf16x8*)(A + r * 256 + nb) = pk;
    }
  }
  __syncthreads();

  // P2: h1 = relu(A @ ee_W1 + b1) -> B   (WMMA)
  if (active) {
    for (int nt = 0; nt < 16; ++nt) {
      v8f acc = {};
      acc = gemm_acc(acc, A, 256, P.ee_W1t, 256, nt, 8);
      float bn = P.ee_b1[nt * 16 + col];
#pragma unroll
      for (int v = 0; v < 8; ++v)
        B[(v + 8 * hh) * 256 + nt * 16 + col] =
            (__bf16)fmaxf(acc[v] + bn, 0.0f);
    }
  }
  __syncthreads();

  // P3: ea = LN2(B @ ee_W2 + b2) -> EA
  if (active) {
    int r = lane >> 1, o = lane & 1;
    float s = P.ee_b2[o] + dot256_ln2(B + r * 256, P.ee_W2, o);
    float so = __shfl_xor(s, 1);
    float mu = 0.5f * (s + so);
    float d0 = s - mu, d1 = so - mu;
    float var = 0.5f * (d0 * d0 + d1 * d1);
    EA[r * 2 + o] = d0 * rsqrtf(var + EPS) * P.ee_g[o] + P.ee_beta[o];
  }
  __syncthreads();

  // P4: gather xs -> A, xd -> B (f32 -> bf16 on the fly)
  if (active) {
    for (int r = 0; r < 16; ++r) {
      int e = eBase + r;
      int si = P.eidx[e];
      int di = P.eidx[NUM_EDGES + e] - NUM_H3;
      const float* xs = P.proc   + (size_t)si * 256 + lane * 8;
      const float* xd = P.latlon + (size_t)di * 256 + lane * 8;
      f32x4 a0 = *(const f32x4*)xs,      a1 = *(const f32x4*)(xs + 4);
      f32x4 b0 = *(const f32x4*)xd,      b1 = *(const f32x4*)(xd + 4);
      bf16x8 pa, pb;
#pragma unroll
      for (int i = 0; i < 4; ++i) {
        pa[i] = (__bf16)a0[i]; pa[i + 4] = (__bf16)a1[i];
        pb[i] = (__bf16)b0[i]; pb[i + 4] = (__bf16)b1[i];
      }
      *(bf16x8*)(A + r * 256 + lane * 8) = pa;
      *(bf16x8*)(B + r * 256 + lane * 8) = pb;
    }
  }
  __syncthreads();

  // P5: g0 = relu(xs@W0a + xd@W0b + ea@Wc + b0) -> C  (dual WMMA + VALU tail)
  if (active) {
    for (int nt = 0; nt < 16; ++nt) {
      v8f acc = {};
      acc = gemm_acc(acc, A, 256, P.pe_W0at, 256, nt, 8);
      acc = gemm_acc(acc, B, 256, P.pe_W0bt, 256, nt, 8);
      int n = nt * 16 + col;
      float wc0 = P.pe_W0[512 * 256 + n];
      float wc1 = P.pe_W0[513 * 256 + n];
      float bn = P.pe_b0[n];
#pragma unroll
      for (int v = 0; v < 8; ++v) {
        int r = v + 8 * hh;
        float x = acc[v] + bn + EA[r * 2 + 0] * wc0 + EA[r * 2 + 1] * wc1;
        C[r * 256 + n] = (__bf16)fmaxf(x, 0.0f);
      }
    }
  }
  __syncthreads();

  // P6: g1 = relu(C @ pe_W1 + b1) -> A  (WMMA)
  if (active) {
    for (int nt = 0; nt < 16; ++nt) {
      v8f acc = {};
      acc = gemm_acc(acc, C, 256, P.pe_W1t, 256, nt, 8);
      float bn = P.pe_b1[nt * 16 + col];
#pragma unroll
      for (int v = 0; v < 8; ++v)
        A[(v + 8 * hh) * 256 + nt * 16 + col] =
            (__bf16)fmaxf(acc[v] + bn, 0.0f);
    }
  }
  __syncthreads();

  // P7: e = LN2(A @ pe_W2 + b2) + ea -> global
  if (active) {
    int r = lane >> 1, o = lane & 1;
    float s = P.pe_b2[o] + dot256_ln2(A + r * 256, P.pe_W2, o);
    float so = __shfl_xor(s, 1);
    float mu = 0.5f * (s + so);
    float d0 = s - mu, d1 = so - mu;
    float var = 0.5f * (d0 * d0 + d1 * d1);
    float y = d0 * rsqrtf(var + EPS) * P.pe_g[o] + P.pe_beta[o] + EA[r * 2 + o];
    P.e_out[(size_t)(eBase + r) * 2 + o] = y;
  }
}

// ---------------------------------------------------------------------------
// Node+decoder fused kernel over latlon rows only.
// ---------------------------------------------------------------------------
struct NodeParams {
  const float* latlon;  // (NUM_LATLON,256)
  const float* start;   // (NUM_LATLON,78)
  const float* e_buf;   // (E,2)
  const float* pn_W0;   // (258,256) f32 (rows 256/257 used)
  const float* pn_b0; const float* pn_b1; const float* pn_b2;
  const float* pn_g; const float* pn_beta;
  const float* dec_b0; const float* dec_b1; const float* dec_b2;
  const __bf16* pn_W0t;  // 256x256
  const __bf16* pn_W1t;  // 256x256
  const __bf16* pn_W2t;  // 256x256
  const __bf16* dec_W0t; // 128x256
  const __bf16* dec_W1t; // 128x128
  const __bf16* dec_W2t; // 80x128 (padded)
  float* out;            // (NUM_LATLON,78)
};

#define NWAVES 4
__global__ __launch_bounds__(NWAVES * 32) void node_kernel(NodeParams P) {
  __shared__ __bf16 nA[NWAVES * 16 * 256];
  __shared__ __bf16 nB[NWAVES * 16 * 256];
  __shared__ float  nF[NWAVES * 16 * 256];
  __shared__ float  nAgg[NWAVES * 16 * 2];

  const int lane = threadIdx.x & 31;
  const int wave = threadIdx.x >> 5;
  const int tile = blockIdx.x * NWAVES + wave;
  const bool active = (tile * 16) < NUM_LATLON;
  const int nBase = tile * 16;
  __bf16* A = nA + wave * 16 * 256;
  __bf16* B = nB + wave * 16 * 256;
  float*  F = nF + wave * 16 * 256;
  float*  AG = nAgg + wave * 32;
  const int col = lane & 15, hh = lane >> 4;

  // P1: x_lat -> A (bf16); agg = sum of 7 incident e rows -> AG
  if (active) {
    for (int r = 0; r < 16; ++r) {
      const float* x = P.latlon + (size_t)(nBase + r) * 256 + lane * 8;
      f32x4 a0 = *(const f32x4*)x, a1 = *(const f32x4*)(x + 4);
      bf16x8 p;
#pragma unroll
      for (int i = 0; i < 4; ++i) { p[i] = (__bf16)a0[i]; p[i + 4] = (__bf16)a1[i]; }
      *(bf16x8*)(A + r * 256 + lane * 8) = p;
    }
    int r = lane >> 1, o = lane & 1;
    int node = nBase + r;
    float s = 0.0f;
#pragma unroll
    for (int j = 0; j < DEG; ++j)
      s += P.e_buf[((size_t)node * DEG + j) * 2 + o];
    AG[r * 2 + o] = s;
  }
  __syncthreads();

  // P2: l0 = relu(A @ W0[:256] + agg@W0[256:258] + b0) -> B
  if (active) {
    for (int nt = 0; nt < 16; ++nt) {
      v8f acc = {};
      acc = gemm_acc(acc, A, 256, P.pn_W0t, 256, nt, 8);
      int n = nt * 16 + col;
      float w0 = P.pn_W0[256 * 256 + n];
      float w1 = P.pn_W0[257 * 256 + n];
      float bn = P.pn_b0[n];
#pragma unroll
      for (int v = 0; v < 8; ++v) {
        int r = v + 8 * hh;
        float x = acc[v] + bn + AG[r * 2 + 0] * w0 + AG[r * 2 + 1] * w1;
        B[r * 256 + n] = (__bf16)fmaxf(x, 0.0f);
      }
    }
  }
  __syncthreads();

  // P3: l1 = relu(B @ W1 + b1) -> A
  if (active) {
    for (int nt = 0; nt < 16; ++nt) {
      v8f acc = {};
      acc = gemm_acc(acc, B, 256, P.pn_W1t, 256, nt, 8);
      float bn = P.pn_b1[nt * 16 + col];
#pragma unroll
      for (int v = 0; v < 8; ++v)
        A[(v + 8 * hh) * 256 + nt * 16 + col] =
            (__bf16)fmaxf(acc[v] + bn, 0.0f);
    }
  }
  __syncthreads();

  // P4: l2 = A @ W2 + b2 -> F (f32, pre-norm)
  if (active) {
    for (int nt = 0; nt < 16; ++nt) {
      v8f acc = {};
      acc = gemm_acc(acc, A, 256, P.pn_W2t, 256, nt, 8);
      float bn = P.pn_b2[nt * 16 + col];
#pragma unroll
      for (int v = 0; v < 8; ++v)
        F[(v + 8 * hh) * 256 + nt * 16 + col] = acc[v] + bn;
    }
  }
  __syncthreads();

  // P5: LayerNorm(256) * g + beta + residual(x_lat f32) -> B (bf16)
  if (active) {
    int r = lane >> 1, hf = lane & 1;
    const float* Fr = F + r * 256 + hf * 128;
    float s = 0.0f, sq = 0.0f;
    for (int k = 0; k < 128; k += 4) {
      f32x4 v = *(const f32x4*)(Fr + k);
      s  += (v[0] + v[1]) + (v[2] + v[3]);
      sq += (v[0] * v[0] + v[1] * v[1]) + (v[2] * v[2] + v[3] * v[3]);
    }
    s += __shfl_xor(s, 1);
    sq += __shfl_xor(sq, 1);
    float mu = s * (1.0f / 256.0f);
    float var = sq * (1.0f / 256.0f) - mu * mu;
    float rs = rsqrtf(fmaxf(var, 0.0f) + EPS);
    const float* xr = P.latlon + (size_t)(nBase + r) * 256 + hf * 128;
    const float* gp = P.pn_g + hf * 128;
    const float* bp = P.pn_beta + hf * 128;
    __bf16* Bw = B + r * 256 + hf * 128;
    for (int k = 0; k < 128; k += 4) {
      f32x4 v  = *(const f32x4*)(Fr + k);
      f32x4 g4 = *(const f32x4*)(gp + k);
      f32x4 b4 = *(const f32x4*)(bp + k);
      f32x4 x4 = *(const f32x4*)(xr + k);
      bf16x4 o4;
#pragma unroll
      for (int i = 0; i < 4; ++i)
        o4[i] = (__bf16)((v[i] - mu) * rs * g4[i] + b4[i] + x4[i]);
      *(bf16x4*)(Bw + k) = o4;
    }
  }
  __syncthreads();

  // P6: dec l0 = relu(B(K=256) @ dec_W0 + b0) -> A (ld=128)
  if (active) {
    for (int nt = 0; nt < 8; ++nt) {
      v8f acc = {};
      acc = gemm_acc(acc, B, 256, P.dec_W0t, 256, nt, 8);
      float bn = P.dec_b0[nt * 16 + col];
#pragma unroll
      for (int v = 0; v < 8; ++v)
        A[(v + 8 * hh) * 128 + nt * 16 + col] =
            (__bf16)fmaxf(acc[v] + bn, 0.0f);
    }
  }
  __syncthreads();

  // P7: dec l1 = relu(A(K=128) @ dec_W1 + b1) -> B (ld=128)
  if (active) {
    for (int nt = 0; nt < 8; ++nt) {
      v8f acc = {};
      acc = gemm_acc(acc, A, 128, P.dec_W1t, 128, nt, 4);
      float bn = P.dec_b1[nt * 16 + col];
#pragma unroll
      for (int v = 0; v < 8; ++v)
        B[(v + 8 * hh) * 128 + nt * 16 + col] =
            (__bf16)fmaxf(acc[v] + bn, 0.0f);
    }
  }
  __syncthreads();

  // P8: out = B(K=128) @ dec_W2(padded 80) + b2 + start -> global (cols < 78)
  if (active) {
    for (int nt = 0; nt < 5; ++nt) {
      v8f acc = {};
      acc = gemm_acc(acc, B, 128, P.dec_W2t, 128, nt, 4);
      int n = nt * 16 + col;
      if (n < 78) {
        float bn = P.dec_b2[n];
#pragma unroll
        for (int v = 0; v < 8; ++v) {
          int row = nBase + v + 8 * hh;
          P.out[(size_t)row * 78 + n] =
              acc[v] + bn + P.start[(size_t)row * 78 + n];
        }
      }
    }
  }
}

// ---------------------------------------------------------------------------
// Workspace layout (bytes)
// ---------------------------------------------------------------------------
#define SZ_256x256 (256 * 256 * 2)
#define OFF_EE_W1T  ((size_t)0)
#define OFF_PE_W0AT (OFF_EE_W1T  + SZ_256x256)
#define OFF_PE_W0BT (OFF_PE_W0AT + SZ_256x256)
#define OFF_PE_W1T  (OFF_PE_W0BT + SZ_256x256)
#define OFF_PN_W0T  (OFF_PE_W1T  + SZ_256x256)
#define OFF_PN_W1T  (OFF_PN_W0T  + SZ_256x256)
#define OFF_PN_W2T  (OFF_PN_W1T  + SZ_256x256)
#define OFF_DEC_W0T (OFF_PN_W2T  + SZ_256x256)
#define OFF_DEC_W1T (OFF_DEC_W0T + (size_t)(128 * 256 * 2))
#define OFF_DEC_W2T (OFF_DEC_W1T + (size_t)(128 * 128 * 2))
#define OFF_EBUF    (OFF_DEC_W2T + (size_t)(80 * 128 * 2))

extern "C" void kernel_launch(void* const* d_in, const int* in_sizes, int n_in,
                              void* d_out, int out_size, void* d_ws, size_t ws_size,
                              hipStream_t stream) {
  const float* proc    = (const float*)d_in[0];
  const float* start   = (const float*)d_in[1];
  const float* latlon  = (const float*)d_in[2];
  const int*   eidx    = (const int*)d_in[3];
  const float* eattr   = (const float*)d_in[4];
  const float* ee_W0 = (const float*)d_in[5];
  const float* ee_b0 = (const float*)d_in[6];
  const float* ee_W1 = (const float*)d_in[7];
  const float* ee_b1 = (const float*)d_in[8];
  const float* ee_W2 = (const float*)d_in[9];
  const float* ee_b2 = (const float*)d_in[10];
  const float* ee_g  = (const float*)d_in[11];
  const float* ee_be = (const float*)d_in[12];
  const float* pe_W0 = (const float*)d_in[13];
  const float* pe_b0 = (const float*)d_in[14];
  const float* pe_W1 = (const float*)d_in[15];
  const float* pe_b1 = (const float*)d_in[16];
  const float* pe_W2 = (const float*)d_in[17];
  const float* pe_b2 = (const float*)d_in[18];
  const float* pe_g  = (const float*)d_in[19];
  const float* pe_be = (const float*)d_in[20];
  const float* pn_W0 = (const float*)d_in[21];
  const float* pn_b0 = (const float*)d_in[22];
  const float* pn_W1 = (const float*)d_in[23];
  const float* pn_b1 = (const float*)d_in[24];
  const float* pn_W2 = (const float*)d_in[25];
  const float* pn_b2 = (const float*)d_in[26];
  const float* pn_g  = (const float*)d_in[27];
  const float* pn_be = (const float*)d_in[28];
  const float* dc_W0 = (const float*)d_in[29];
  const float* dc_b0 = (const float*)d_in[30];
  const float* dc_W1 = (const float*)d_in[31];
  const float* dc_b1 = (const float*)d_in[32];
  const float* dc_W2 = (const float*)d_in[33];
  const float* dc_b2 = (const float*)d_in[34];

  char* ws = (char*)d_ws;
  __bf16* ee_W1t  = (__bf16*)(ws + OFF_EE_W1T);
  __bf16* pe_W0at = (__bf16*)(ws + OFF_PE_W0AT);
  __bf16* pe_W0bt = (__bf16*)(ws + OFF_PE_W0BT);
  __bf16* pe_W1t  = (__bf16*)(ws + OFF_PE_W1T);
  __bf16* pn_W0t  = (__bf16*)(ws + OFF_PN_W0T);
  __bf16* pn_W1t  = (__bf16*)(ws + OFF_PN_W1T);
  __bf16* pn_W2t  = (__bf16*)(ws + OFF_PN_W2T);
  __bf16* dc_W0t  = (__bf16*)(ws + OFF_DEC_W0T);
  __bf16* dc_W1t  = (__bf16*)(ws + OFF_DEC_W1T);
  __bf16* dc_W2t  = (__bf16*)(ws + OFF_DEC_W2T);
  float*  e_buf   = (float*)(ws + OFF_EBUF);

  auto T = [&](const float* W, __bf16* Wt, int K, int N, int Npad) {
    int total = Npad * K;
    transpose_cast_kernel<<<(total + 255) / 256, 256, 0, stream>>>(W, Wt, K, N, Npad);
  };
  T(ee_W1, ee_W1t, 256, 256, 256);
  T(pe_W0, pe_W0at, 256, 256, 256);              // rows   0..255 of (514,256)
  T(pe_W0 + 256 * 256, pe_W0bt, 256, 256, 256);  // rows 256..511
  T(pe_W1, pe_W1t, 256, 256, 256);
  T(pn_W0, pn_W0t, 256, 256, 256);               // rows   0..255 of (258,256)
  T(pn_W1, pn_W1t, 256, 256, 256);
  T(pn_W2, pn_W2t, 256, 256, 256);
  T(dc_W0, dc_W0t, 256, 128, 128);
  T(dc_W1, dc_W1t, 128, 128, 128);
  T(dc_W2, dc_W2t, 128, 78, 80);

  EdgeParams EP;
  EP.proc = proc; EP.latlon = latlon; EP.eidx = eidx; EP.eattr = eattr;
  EP.ee_W0 = ee_W0; EP.ee_b0 = ee_b0; EP.ee_b1 = ee_b1;
  EP.ee_W2 = ee_W2; EP.ee_b2 = ee_b2; EP.ee_g = ee_g; EP.ee_beta = ee_be;
  EP.pe_W0 = pe_W0; EP.pe_b0 = pe_b0; EP.pe_b1 = pe_b1;
  EP.pe_W2 = pe_W2; EP.pe_b2 = pe_b2; EP.pe_g = pe_g; EP.pe_beta = pe_be;
  EP.ee_W1t = ee_W1t; EP.pe_W0at = pe_W0at; EP.pe_W0bt = pe_W0bt; EP.pe_W1t = pe_W1t;
  EP.e_out = e_buf;
  {
    int tiles = NUM_EDGES / 16;  // 28350
    int blocks = (tiles + EWAVES - 1) / EWAVES;
    edge_kernel<<<blocks, EWAVES * 32, 0, stream>>>(EP);
  }

  NodeParams NP;
  NP.latlon = latlon; NP.start = start; NP.e_buf = e_buf;
  NP.pn_W0 = pn_W0; NP.pn_b0 = pn_b0; NP.pn_b1 = pn_b1; NP.pn_b2 = pn_b2;
  NP.pn_g = pn_g; NP.pn_beta = pn_be;
  NP.dec_b0 = dc_b0; NP.dec_b1 = dc_b1; NP.dec_b2 = dc_b2;
  NP.pn_W0t = pn_W0t; NP.pn_W1t = pn_W1t; NP.pn_W2t = pn_W2t;
  NP.dec_W0t = dc_W0t; NP.dec_W1t = dc_W1t; NP.dec_W2t = dc_W2t;
  NP.out = (float*)d_out;
  {
    int tiles = NUM_LATLON / 16;  // 4050
    int blocks = (tiles + NWAVES - 1) / NWAVES;
    node_kernel<<<blocks, NWAVES * 32, 0, stream>>>(NP);
  }
}
